// OFDMGeneratorModule_53644141527078
// MI455X (gfx1250) — matrix operand order repaired and checked by reading, not compiled, not run on hardware
//
#include <hip/hip_runtime.h>

typedef _Float16 v16h __attribute__((ext_vector_type(16)));
typedef _Float16 v8h  __attribute__((ext_vector_type(8)));
typedef float    v8f  __attribute__((ext_vector_type(8)));
typedef float    v4f  __attribute__((ext_vector_type(4)));

#define LDS_LANE_STRIDE 24              // 16 halves used + pad -> 48B lane stride (conflict-free b128)
#define LDS_ARR_STRIDE  (8 * 32 * LDS_LANE_STRIDE)   // 8 frags per array

static __device__ __forceinline__ v16h ld_frag16(const _Float16* p) {
    // 16B-aligned load of 16 halves as two v8h (avoids v16h 32B-alignment assumption)
    v8h a = *(const v8h*)p;
    v8h b = *(const v8h*)(p + 8);
    return __builtin_shufflevector(a, b, 0,1,2,3,4,5,6,7,8,9,10,11,12,13,14,15);
}

static __device__ __forceinline__ v8f wmma16(v16h a, v16h b, v8f c) {
    return __builtin_amdgcn_wmma_f32_16x16x32_f16(false, a, false, b, (short)0, c, false, false);
}

// Out(B x 64) = X(B x 64) @ K^T via f32 = (hi+lo) f16 split, 16x16x32 f16 WMMA.
// A-frag (16x32 f16, ISA table): lane l=h*16+m, element e: K = 32c + 16*(e>>3) + 8h + (e&7)
// B-frag (32x16 f16, ISA table): lane l=h*16+n, element e: K = 32c + 16*(e>>3) + 8h + (e&7), N = l%16
// C/D (16x16 f32): lane l, vgpr j -> M = j + 8h, N = l%16
__global__ __launch_bounds__(256) void ofdm_idft_wmma_f16split(
    const float* __restrict__ x,     // [B][2][64]
    const float* __restrict__ cosk,  // [64][64]
    const float* __restrict__ sink,  // [64][64]
    float* __restrict__ out,         // [B][2][64]
    int batch)
{
    // B fragments in LDS: [arr: cosHi,cosLo,sinHi,sinLo][frag f = nt*2 + c][lane][24 halves]
    __shared__ _Float16 lds[4 * LDS_ARR_STRIDE];   // 48 KB

    const int tid = threadIdx.x;

    // ---- One-time: split kernels into f16 hi/lo and repack into B-fragment order ----
    for (int i = tid; i < 4096; i += 256) {
        const int n = i >> 6;
        const int k = i & 63;
        const int c  = k >> 5;          // K-chunk (0/1)
        const int hh = (k >> 3) & 1;    // half-wave
        const int e  = ((k >> 4) & 1) * 8 + (k & 7);   // element within fragment
        const int l  = hh * 16 + (n & 15);
        const int f  = (n >> 4) * 2 + c;
        const int base = (f * 32 + l) * LDS_LANE_STRIDE + e;

        const float cv = cosk[i];
        const float sv = sink[i];
        const _Float16 chi = (_Float16)cv;
        const _Float16 clo = (_Float16)(cv - (float)chi);
        const _Float16 shi = (_Float16)sv;
        const _Float16 slo = (_Float16)(sv - (float)shi);
        lds[0 * LDS_ARR_STRIDE + base] = chi;
        lds[1 * LDS_ARR_STRIDE + base] = clo;
        lds[2 * LDS_ARR_STRIDE + base] = shi;
        lds[3 * LDS_ARR_STRIDE + base] = slo;
    }
    __syncthreads();

    const int lane = tid & 31;
    const int wave = tid >> 5;
    const int h    = lane >> 4;
    const int m    = lane & 15;

    const int tile = blockIdx.x * 8 + wave;   // 16 batch rows per wave
    const long b0  = (long)tile * 16;
    if (b0 >= batch) return;                  // wave-uniform; EXEC stays all-ones

    // ---- Load X rows, patch pilots, split into f16 hi/lo A-fragments ----
    v16h arhi[2], arlo[2], aihi[2], ailo[2];
    const float* rowr = x + (b0 + m) * 128;
    const float* rowi = rowr + 64;
#pragma unroll
    for (int c = 0; c < 2; ++c) {
        float gr[16], gi[16];
#pragma unroll
        for (int g = 0; g < 2; ++g) {
            const int kb = 32 * c + 16 * g + 8 * h;     // 32B-aligned within row
            const v4f r0 = *(const v4f*)(rowr + kb);
            const v4f r1 = *(const v4f*)(rowr + kb + 4);
            const v4f i0 = *(const v4f*)(rowi + kb);
            const v4f i1 = *(const v4f*)(rowi + kb + 4);
#pragma unroll
            for (int t = 0; t < 4; ++t) {
                gr[g * 8 + t]     = r0[t];
                gr[g * 8 + 4 + t] = r1[t];
                gi[g * 8 + t]     = i0[t];
                gi[g * 8 + 4 + t] = i1[t];
            }
        }
#pragma unroll
        for (int e = 0; e < 16; ++e) {
            const int kk = 32 * c + 16 * (e >> 3) + 8 * h + (e & 7);
            const bool pilot = (kk == 11) || (kk == 25) || (kk == 39) || (kk == 53);
            const float xr = pilot ? 1.0f : gr[e];
            const float xi = pilot ? 0.0f : gi[e];
            const _Float16 rh = (_Float16)xr;
            const _Float16 ih = (_Float16)xi;
            arhi[c][e] = rh;  arlo[c][e] = (_Float16)(xr - (float)rh);
            aihi[c][e] = ih;  ailo[c][e] = (_Float16)(xi - (float)ih);
        }
    }

    // ---- 4 N-tiles; per tile 2 K-chunks x 12 WMMAs (3-product split x 4 matmuls) ----
#pragma unroll
    for (int nt = 0; nt < 4; ++nt) {
        v8f crr = {};  // Xr @ C^T
        v8f cri = {};  // Xi @ C^T
        v8f cir = {};  // Xr @ S^T
        v8f cii = {};  // Xi @ S^T
#pragma unroll
        for (int c = 0; c < 2; ++c) {
            const int f = nt * 2 + c;
            const _Float16* bb = lds + (f * 32 + lane) * LDS_LANE_STRIDE;
            const v16h bchi = ld_frag16(bb + 0 * LDS_ARR_STRIDE);
            const v16h bclo = ld_frag16(bb + 1 * LDS_ARR_STRIDE);
            const v16h bshi = ld_frag16(bb + 2 * LDS_ARR_STRIDE);
            const v16h bslo = ld_frag16(bb + 3 * LDS_ARR_STRIDE);
            // hi*hi
            crr = wmma16(arhi[c], bchi, crr);
            cri = wmma16(aihi[c], bchi, cri);
            cir = wmma16(arhi[c], bshi, cir);
            cii = wmma16(aihi[c], bshi, cii);
            // hi*lo
            crr = wmma16(arhi[c], bclo, crr);
            cri = wmma16(aihi[c], bclo, cri);
            cir = wmma16(arhi[c], bslo, cir);
            cii = wmma16(aihi[c], bslo, cii);
            // lo*hi
            crr = wmma16(arlo[c], bchi, crr);
            cri = wmma16(ailo[c], bchi, cri);
            cir = wmma16(arlo[c], bshi, cir);
            cii = wmma16(ailo[c], bshi, cii);
        }
        // out_real = crr - cii ; out_imag = cri + cir
        float* obase = out + b0 * 128 + nt * 16 + m;
#pragma unroll
        for (int j = 0; j < 8; ++j) {
            const int brow = j + 8 * h;
            obase[brow * 128 + 0]  = crr[j] - cii[j];
            obase[brow * 128 + 64] = cri[j] + cir[j];
        }
    }
}

extern "C" void kernel_launch(void* const* d_in, const int* in_sizes, int n_in,
                              void* d_out, int out_size, void* d_ws, size_t ws_size,
                              hipStream_t stream) {
    const float* x    = (const float*)d_in[0];  // eq_freq  [B][2][64] f32
    const float* cosk = (const float*)d_in[1];  // [64][64] f32
    const float* sink = (const float*)d_in[2];  // [64][64] f32
    float* out = (float*)d_out;                 // [B][2][64] f32

    const int batch  = in_sizes[0] / 128;
    const int tiles  = (batch + 15) / 16;
    const int blocks = (tiles + 7) / 8;
    ofdm_idft_wmma_f16split<<<blocks, 256, 0, stream>>>(x, cosk, sink, out, batch);
}
